// LocalSlidingWindow_DisentangledAttention_49993419325576
// MI455X (gfx1250) — compile-verified, matrix operand order checked by reading, not run on hardware
//
#include <hip/hip_runtime.h>
#include <cstddef>
#include <cstdint>

// ---------------- constants ----------------
#define BB   4
#define SS   8192
#define HH   768
#define NHH  12
#define HDD  64
#define BSZ  128          // sliding window block size
#define NBLK (SS / BSZ)   // 64
#define MQ   32           // queries per attention workgroup
#define PBUCK 512         // 2*BUCKETS

typedef __bf16 bf16_t;
typedef __attribute__((ext_vector_type(16))) bf16_t v16bf;
typedef __attribute__((ext_vector_type(8)))  float  v8f;
typedef __attribute__((ext_vector_type(4)))  unsigned int v4u;
typedef __attribute__((ext_vector_type(8)))  int v8i;
typedef __attribute__((ext_vector_type(4)))  int v4i;

#if __has_builtin(__builtin_amdgcn_tensor_load_to_lds)
#define HAVE_TDM 1
#else
#define HAVE_TDM 0
#endif

union FragB {
    v16bf v;
    uint4 q[2];
    unsigned short u[16];
};

// fp32 -> bf16 (round to nearest even)
__device__ __forceinline__ unsigned short f2bf(float x) {
    unsigned u = __float_as_uint(x);
    unsigned r = u + 0x7FFFu + ((u >> 16) & 1u);
    return (unsigned short)(r >> 16);
}

__device__ __forceinline__ v8f wmma_bf16(v16bf a, v16bf b, v8f c) {
    return __builtin_amdgcn_wmma_f32_16x16x32_bf16(false, a, false, b, (short)0, c, false, false);
}

// Fragment with contraction over the contiguous (last) dim of a row-major
// [rows][ld] bf16 LDS tile -> 2x ds_load_b128 (ld % 8 == 0, k0 % 32 == 0).
__device__ __forceinline__ v16bf load_frag_k(const unsigned short* s, int r0, int k0, int ld) {
    int lane = threadIdx.x & 31;
    int l = lane & 15, hi = lane >> 4;
    const unsigned short* p = s + (size_t)(r0 + l) * ld + k0 + 8 * hi;
    FragB f;
    f.q[0] = *(const uint4*)p;
    f.q[1] = *(const uint4*)(p + 16);
    return f.v;
}

// Same but fp32 source converted to bf16 in flight (probs @ V). 4x ds_load_b128.
__device__ __forceinline__ v16bf load_frag_k_f32(const float* s, int r0, int k0, int ld) {
    int lane = threadIdx.x & 31;
    int l = lane & 15, hi = lane >> 4;
    const float* p = s + (size_t)(r0 + l) * ld + k0 + 8 * hi;
    float4 a0 = *(const float4*)(p);
    float4 a1 = *(const float4*)(p + 4);
    float4 b0 = *(const float4*)(p + 16);
    float4 b1 = *(const float4*)(p + 20);
    FragB f;
    f.u[0]  = f2bf(a0.x); f.u[1]  = f2bf(a0.y); f.u[2]  = f2bf(a0.z); f.u[3]  = f2bf(a0.w);
    f.u[4]  = f2bf(a1.x); f.u[5]  = f2bf(a1.y); f.u[6]  = f2bf(a1.z); f.u[7]  = f2bf(a1.w);
    f.u[8]  = f2bf(b0.x); f.u[9]  = f2bf(b0.y); f.u[10] = f2bf(b0.z); f.u[11] = f2bf(b0.w);
    f.u[12] = f2bf(b1.x); f.u[13] = f2bf(b1.y); f.u[14] = f2bf(b1.z); f.u[15] = f2bf(b1.w);
    return f.v;
}

// C/D 16x16 f32 store: VGPR r -> M = r + 8*hi, N = lane&15.
__device__ __forceinline__ void store_c(float* s, int r0, int c0, int ld, v8f acc) {
    int lane = threadIdx.x & 31;
    int n = lane & 15, hi = lane >> 4;
#pragma unroll
    for (int r = 0; r < 8; ++r) s[(size_t)(r0 + r + 8 * hi) * ld + c0 + n] = acc[r];
}

#if HAVE_TDM
// TDM 2D tile load: rows_tile rows of 64 contiguous bf16, global row stride
// `stride` elements, into LDS with 8-bf16 padding every 64 (pad_interval=32
// dwords, pad_amount=4 dwords -> LDS row stride 72 ushorts). Rows >= rows_valid
// are zero-filled by TDM OOB handling (tensor_dim1 = rows_valid).
// D# layout per CDNA5 ISA 8.3/8.4; 6-arg builtin (v4u, v8i, v4i, v4i, v8i, cpol).
__device__ __forceinline__ void tdm_load_2d(unsigned lds_byte_off,
                                            const unsigned short* gbase,
                                            int rows_tile, int rows_valid, int stride) {
    unsigned long long ga = (unsigned long long)(uintptr_t)gbase;
    v4u g0;
    g0[0] = 1u;                                                  // count=1 (valid user D#)
    g0[1] = lds_byte_off;                                        // lds_addr
    g0[2] = (unsigned)ga;                                        // global_addr[31:0]
    g0[3] = ((unsigned)(ga >> 32) & 0x01FFFFFFu) | (2u << 30);   // global_addr[56:32], type=2
    v8i g1;
    g1[0] = (int)((1u << 16) |     // data_size = 2 bytes
                  (1u << 20) |     // pad_enable
                  (4u << 22) |     // pad_interval: 32 dwords (64 bf16)
                  (3u << 25));     // pad_amount: 4 dwords (8 bf16)
    g1[1] = (int)(64u << 16);                                    // tensor_dim0 = 64 (bits 79:48)
    g1[2] = (int)(((unsigned)rows_valid & 0xFFFFu) << 16);       // tensor_dim1 (bits 111:80)
    g1[3] = (int)(64u << 16);                                    // tile_dim0 = 64 (bits 127:112)
    g1[4] = (int)((unsigned)rows_tile & 0xFFFFu);                // tile_dim1 (bits 143:128)
    g1[5] = stride;                                              // tensor_dim0_stride (bits 207:160)
    g1[6] = 0;
    g1[7] = 0;
    v4i z4; z4[0] = 0; z4[1] = 0; z4[2] = 0; z4[3] = 0;
    v8i z8; z8[0] = 0; z8[1] = 0; z8[2] = 0; z8[3] = 0;
    z8[4] = 0; z8[5] = 0; z8[6] = 0; z8[7] = 0;
    __builtin_amdgcn_tensor_load_to_lds(g0, g1, z4, z4, z8, 0);
}
__device__ __forceinline__ unsigned lds_off(const void* p) {
    return (unsigned)(uintptr_t)p;   // LDS aperture: low 32 bits = LDS byte address
}
#endif

// DeBERTa log-bucket index for relative position d = qpos - kpos, buckets=256, mid=128.
__device__ __forceinline__ int bucket_idx(int d) {
    if (d >= -128 && d <= 128) {
        int idx = d + 256;
        return idx < 0 ? 0 : (idx > 511 ? 511 : idx);
    }
    float a  = fabsf((float)d);
    float lp = ceilf(logf(a * (1.0f / 128.0f)) * (127.0f / 0.6892332f)) + 128.0f;
    int r = (d < 0) ? -(int)lp : (int)lp;
    int idx = r + 256;
    return idx < 0 ? 0 : (idx > 511 ? 511 : idx);
}

// ---------------- fp32 -> bf16 conversion (float4 in, uint2 out) ----------------
__global__ __launch_bounds__(256) void cvt_kernel(const float* __restrict__ src,
                                                  unsigned short* __restrict__ dst, int n4) {
    for (int i = blockIdx.x * 256 + threadIdx.x; i < n4; i += gridDim.x * 256) {
        float4 v = ((const float4*)src)[i];
        uint2 o;
        o.x = (unsigned)f2bf(v.x) | ((unsigned)f2bf(v.y) << 16);
        o.y = (unsigned)f2bf(v.z) | ((unsigned)f2bf(v.w) << 16);
        ((uint2*)dst)[i] = o;
    }
}

// ---------------- weight transpose + fp32 -> bf16: dst[n][k] = src[k][n], 768x768 ----------------
__global__ __launch_bounds__(256) void cvt_w_kernel(const float* __restrict__ src,
                                                    unsigned short* __restrict__ dst) {
    for (int i = blockIdx.x * 256 + threadIdx.x; i < HH * HH; i += gridDim.x * 256) {
        int n = i / HH, k = i - n * HH;
        dst[i] = f2bf(src[(size_t)k * HH + n]);
    }
}

// ---------------- GEMM: C[M,N] = A_bf[M,K] @ Wt_bf[N,K]^T + bias ----------------
// 256 threads (8 waves), tile 128(M) x 64(N). K chunks of 64, TDM-double-buffered:
// wave 0 issues tensor_load_to_lds for chunk c+1 while all waves run WMMAs on
// chunk c. No VGPR staging -> no spills. Weights arrive pre-transposed [N][K].
__global__ __launch_bounds__(256, 1) void gemm_bias_kernel(const unsigned short* __restrict__ A,
                                                           const unsigned short* __restrict__ Wt,
                                                           const float* __restrict__ bias,
                                                           float* __restrict__ Cf,
                                                           unsigned short* __restrict__ Cb,
                                                           int M, int N, int K, int out_f32) {
    __shared__ __align__(16) unsigned short a_s[2][128 * 72];  // [m][k]
    __shared__ __align__(16) unsigned short w_s[2][64 * 72];   // [n][k]

    const int m0 = blockIdx.x * 128;
    const int n0 = blockIdx.y * 64;
    const int wid = threadIdx.x >> 5;
    const int nch = K / 64;

    v8f acc[4];
#pragma unroll
    for (int t = 0; t < 4; ++t)
#pragma unroll
        for (int r = 0; r < 8; ++r) acc[t][r] = 0.0f;

#if HAVE_TDM
    if (wid == 0) {   // prologue: chunk 0 into buffer 0
        tdm_load_2d(lds_off(a_s[0]), A + (size_t)m0 * K, 128, 128, K);
        tdm_load_2d(lds_off(w_s[0]), Wt + (size_t)n0 * K, 64, 64, K);
    }
#endif
    for (int c = 0; c < nch; ++c) {
        int cur;
#if HAVE_TDM
        cur = c & 1;
        if (wid == 0) {
            if (c + 1 < nch) {   // prefetch next chunk into back buffer (async)
                int kc = (c + 1) * 64;
                tdm_load_2d(lds_off(a_s[cur ^ 1]), A + (size_t)m0 * K + kc, 128, 128, K);
                tdm_load_2d(lds_off(w_s[cur ^ 1]), Wt + (size_t)n0 * K + kc, 64, 64, K);
                __builtin_amdgcn_s_wait_tensorcnt(2);   // in-order: chunk c has landed
            } else {
                __builtin_amdgcn_s_wait_tensorcnt(0);
            }
        }
        __syncthreads();
#else
        cur = 0;
        for (int i = threadIdx.x; i < 128 * 8; i += 256) {
            int row = i >> 3, col8 = (i & 7) * 8;
            *(uint4*)&a_s[0][row * 72 + col8] =
                *(const uint4*)&A[(size_t)(m0 + row) * K + c * 64 + col8];
        }
        for (int i = threadIdx.x; i < 64 * 8; i += 256) {
            int row = i >> 3, col8 = (i & 7) * 8;
            *(uint4*)&w_s[0][row * 72 + col8] =
                *(const uint4*)&Wt[(size_t)(n0 + row) * K + c * 64 + col8];
        }
        __syncthreads();
#endif
#pragma unroll
        for (int kk = 0; kk < 64; kk += 32) {
            v16bf af = load_frag_k(a_s[cur], wid * 16, kk, 72);
#pragma unroll
            for (int nt = 0; nt < 4; ++nt) {
                v16bf bf = load_frag_k(w_s[cur], nt * 16, kk, 72);
                acc[nt] = wmma_bf16(af, bf, acc[nt]);
            }
        }
        __syncthreads();
    }

    const int lane = threadIdx.x & 31;
    const int nn = lane & 15, hi = lane >> 4;
#pragma unroll
    for (int nt = 0; nt < 4; ++nt)
#pragma unroll
        for (int r = 0; r < 8; ++r) {
            int row = m0 + wid * 16 + r + 8 * hi;
            int col = n0 + nt * 16 + nn;
            float v = acc[nt][r] + bias[col];
            if (out_f32) Cf[(size_t)row * N + col] = v;
            else         Cb[(size_t)row * N + col] = f2bf(v);
        }
}

// ---------------- fused sliding-window disentangled attention ----------------
// One workgroup per (b, block, 32-query sub-block, head). 256 threads = 8 waves.
__global__ __launch_bounds__(256, 1) void attn_kernel(const unsigned short* __restrict__ qg,
                                                      const unsigned short* __restrict__ kg,
                                                      const unsigned short* __restrict__ vg,
                                                      const unsigned short* __restrict__ poskg,
                                                      const unsigned short* __restrict__ posqg,
                                                      unsigned short* __restrict__ ctxg) {
    __shared__ __align__(16) unsigned short q_s[MQ * 72];    //  4.5 KB  [q][d]
    __shared__ __align__(16) unsigned short kv_s[384 * 72];  // 54.0 KB  k:[k][d], later v^T:[d][k] s=392
    __shared__ __align__(16) unsigned short pos_s[128 * 72]; // 18.0 KB  [p][d]
    __shared__ __align__(16) float sc_s[MQ * 388];           // 48.5 KB  scores 32x384 (+pad)
    __shared__ __align__(16) float stg_s[64 * 132];          // 33.0 KB  wmma staging
    __shared__ short idxd_s[512];                            //  1.0 KB  bucket index by (q-k)

    int id = blockIdx.x;
    const int h   = id % NHH;          id /= NHH;
    const int sub = id & 3;            id >>= 2;
    const int blk = id % NBLK;         id /= NBLK;
    const int b   = id;

    const int s0    = blk * BSZ + sub * MQ;   // first query row (global)
    const int qpos0 = sub * MQ;               // position within 128-block
    const int ks0   = blk * BSZ - BSZ;        // window start (may be negative)
    const int wid   = threadIdx.x >> 5;
    const int lane  = threadIdx.x & 31;

    // relative-position bucket table: d in [-383, 127]
    for (int d = threadIdx.x; d < 511; d += 256)
        idxd_s[d] = (short)bucket_idx(d - 383);

#if HAVE_TDM
    {
        const int lead = (ks0 < 0) ? -ks0 : 0;     // 128 for first block
        if (lead) {                                 // pre-zero padded leading rows
            uint4 zz; zz.x = 0u; zz.y = 0u; zz.z = 0u; zz.w = 0u;
            for (int i = threadIdx.x; i < lead * 9; i += 256)   // 72 ushorts = 9 uint4
                ((uint4*)kv_s)[i] = zz;
        }
        if (wid == 0) {
            const int r0 = ks0 + lead;
            tdm_load_2d(lds_off(q_s),
                        qg + ((size_t)(b * SS + s0) * NHH + h) * HDD, MQ, SS - s0, HH);
            tdm_load_2d(lds_off(kv_s) + (unsigned)(lead * 144),
                        kg + ((size_t)(b * SS + r0) * NHH + h) * HDD, 384 - lead, SS - r0, HH);
            __builtin_amdgcn_s_wait_tensorcnt(0);
        }
    }
#else
    for (int i = threadIdx.x; i < MQ * 8; i += 256) {
        int row = i >> 3, col8 = (i & 7) * 8;
        *(uint4*)&q_s[row * 72 + col8] =
            *(const uint4*)&qg[((size_t)(b * SS + s0 + row) * NHH + h) * HDD + col8];
    }
    for (int i = threadIdx.x; i < 384 * 8; i += 256) {
        int row = i >> 3, col8 = (i & 7) * 8;
        int s = ks0 + row;
        uint4 t; t.x = 0u; t.y = 0u; t.z = 0u; t.w = 0u;
        if (s >= 0 && s < SS)
            t = *(const uint4*)&kg[((size_t)(b * SS + s) * NHH + h) * HDD + col8];
        *(uint4*)&kv_s[row * 72 + col8] = t;
    }
#endif
    __syncthreads();

    // ---- c2c: scores = q @ k^T  (32 x 384), 48 tiles over 8 waves ----
    for (int t = wid; t < 48; t += 8) {
        int tm = t & 1, tn = t >> 1;
        v8f acc;
#pragma unroll
        for (int r = 0; r < 8; ++r) acc[r] = 0.0f;
#pragma unroll
        for (int kk = 0; kk < 64; kk += 32) {
            v16bf af = load_frag_k(q_s, tm * 16, kk, 72);
            v16bf bf = load_frag_k(kv_s, tn * 16, kk, 72);
            acc = wmma_bf16(af, bf, acc);
        }
        store_c(sc_s, tm * 16, tn * 16, 388, acc);
    }
    __syncthreads();

    // ---- p2c: scores[q,k] += k[k] . pos_q[idx(q-k)] ----
    for (int pc = 0; pc < 4; ++pc) {
#if HAVE_TDM
        if (wid == 0) {
            tdm_load_2d(lds_off(pos_s),
                        posqg + ((size_t)(pc * 128) * NHH + h) * HDD, 128, PBUCK - pc * 128, HH);
            __builtin_amdgcn_s_wait_tensorcnt(0);
        }
#else
        for (int i = threadIdx.x; i < 128 * 8; i += 256) {
            int row = i >> 3, col8 = (i & 7) * 8;
            *(uint4*)&pos_s[row * 72 + col8] =
                *(const uint4*)&posqg[((size_t)(pc * 128 + row) * NHH + h) * HDD + col8];
        }
#endif
        __syncthreads();
        for (int ksub = 0; ksub < 6; ++ksub) {
            // stage: (64 k-rows) x (128 buckets) = k_sub @ pos_chunk^T
            for (int t = wid; t < 32; t += 8) {
                int tm = t & 3, tn = t >> 2;
                v8f acc;
#pragma unroll
                for (int r = 0; r < 8; ++r) acc[r] = 0.0f;
#pragma unroll
                for (int kk = 0; kk < 64; kk += 32) {
                    v16bf af = load_frag_k(kv_s, ksub * 64 + tm * 16, kk, 72);
                    v16bf bf = load_frag_k(pos_s, tn * 16, kk, 72);
                    acc = wmma_bf16(af, bf, acc);
                }
                store_c(stg_s, tm * 16, tn * 16, 132, acc);
            }
            __syncthreads();
            for (int i = threadIdx.x; i < MQ * 64; i += 256) {
                int q = i >> 6, krel = i & 63;
                int k = ksub * 64 + krel;
                int idx = idxd_s[qpos0 + q - k + 383];
                if ((idx >> 7) == pc)
                    sc_s[q * 388 + k] += stg_s[krel * 132 + (idx & 127)];
            }
            __syncthreads();
        }
    }

    // ---- c2p: scores[q,k] += q[q] . pos_k[idx(q-k)] ----
    for (int pc = 0; pc < 4; ++pc) {
#if HAVE_TDM
        if (wid == 0) {
            tdm_load_2d(lds_off(pos_s),
                        poskg + ((size_t)(pc * 128) * NHH + h) * HDD, 128, PBUCK - pc * 128, HH);
            __builtin_amdgcn_s_wait_tensorcnt(0);
        }
#else
        for (int i = threadIdx.x; i < 128 * 8; i += 256) {
            int row = i >> 3, col8 = (i & 7) * 8;
            *(uint4*)&pos_s[row * 72 + col8] =
                *(const uint4*)&poskg[((size_t)(pc * 128 + row) * NHH + h) * HDD + col8];
        }
#endif
        __syncthreads();
        for (int t = wid; t < 16; t += 8) {
            int tm = t & 1, tn = t >> 1;
            v8f acc;
#pragma unroll
            for (int r = 0; r < 8; ++r) acc[r] = 0.0f;
#pragma unroll
            for (int kk = 0; kk < 64; kk += 32) {
                v16bf af = load_frag_k(q_s, tm * 16, kk, 72);
                v16bf bf = load_frag_k(pos_s, tn * 16, kk, 72);
                acc = wmma_bf16(af, bf, acc);
            }
            store_c(stg_s, tm * 16, tn * 16, 132, acc);
        }
        __syncthreads();
        for (int i = threadIdx.x; i < MQ * 384; i += 256) {
            int q = i / 384, k = i - q * 384;
            int idx = idxd_s[qpos0 + q - k + 383];
            if ((idx >> 7) == pc)
                sc_s[q * 388 + k] += stg_s[q * 132 + (idx & 127)];
        }
        __syncthreads();
    }

    // ---- softmax over 384 keys (scale = 1/sqrt(3*64)); 8 lanes per row ----
    {
        const float scale = 0.07216878365f;
        int row = threadIdx.x >> 3;
        int l8  = threadIdx.x & 7;
        float m = -3.4e38f;
        for (int j = l8; j < 384; j += 8) m = fmaxf(m, sc_s[row * 388 + j]);
#pragma unroll
        for (int off = 1; off < 8; off <<= 1) m = fmaxf(m, __shfl_xor(m, off, 32));
        float sum = 0.0f;
        for (int j = l8; j < 384; j += 8) {
            float e = __expf((sc_s[row * 388 + j] - m) * scale);
            sc_s[row * 388 + j] = e;
            sum += e;
        }
#pragma unroll
        for (int off = 1; off < 8; off <<= 1) sum += __shfl_xor(sum, off, 32);
        float inv = 1.0f / sum;
        for (int j = l8; j < 384; j += 8) sc_s[row * 388 + j] *= inv;
    }
    __syncthreads();

    // ---- load v window TRANSPOSED into kv_s as [d][k], stride 392 ----
    for (int i = threadIdx.x; i < 384 * 8; i += 256) {
        int row = i >> 3, col8 = (i & 7) * 8;    // row = k, col8 = d
        int s = ks0 + row;
        uint4 t; t.x = 0u; t.y = 0u; t.z = 0u; t.w = 0u;
        if (s >= 0 && s < SS)
            t = *(const uint4*)&vg[((size_t)(b * SS + s) * NHH + h) * HDD + col8];
        kv_s[(col8 + 0) * 392 + row] = (unsigned short)(t.x & 0xFFFFu);
        kv_s[(col8 + 1) * 392 + row] = (unsigned short)(t.x >> 16);
        kv_s[(col8 + 2) * 392 + row] = (unsigned short)(t.y & 0xFFFFu);
        kv_s[(col8 + 3) * 392 + row] = (unsigned short)(t.y >> 16);
        kv_s[(col8 + 4) * 392 + row] = (unsigned short)(t.z & 0xFFFFu);
        kv_s[(col8 + 5) * 392 + row] = (unsigned short)(t.z >> 16);
        kv_s[(col8 + 6) * 392 + row] = (unsigned short)(t.w & 0xFFFFu);
        kv_s[(col8 + 7) * 392 + row] = (unsigned short)(t.w >> 16);
    }
    __syncthreads();

    // ---- ctx = probs @ v (32 x 64); one 16x16 tile per wave ----
    {
        int tm = wid & 1, tn = wid >> 1;
        v8f acc;
#pragma unroll
        for (int r = 0; r < 8; ++r) acc[r] = 0.0f;
        for (int kk = 0; kk < 384; kk += 32) {
            v16bf af = load_frag_k_f32(sc_s, tm * 16, kk, 388);
            v16bf bf = load_frag_k(kv_s, tn * 16, kk, 392);   // v^T: [d][k]
            acc = wmma_bf16(af, bf, acc);
        }
        int nn = lane & 15, hi = lane >> 4;
#pragma unroll
        for (int r = 0; r < 8; ++r) {
            int row = tm * 16 + r + 8 * hi;
            int col = tn * 16 + nn;
            ctxg[((size_t)(b * SS + s0 + row) * NHH + h) * HDD + col] = f2bf(acc[r]);
        }
    }
}

// ---------------- residual + LayerNorm (in place on d_out) ----------------
__global__ __launch_bounds__(256) void ln_kernel(const float* __restrict__ hidden,
                                                 const float* __restrict__ ln_scale,
                                                 const float* __restrict__ ln_bias,
                                                 float* __restrict__ out) {
    int row  = blockIdx.x * 8 + (threadIdx.x >> 5);
    int lane = threadIdx.x & 31;
    float x[24];
    float sum = 0.0f, sq = 0.0f;
#pragma unroll
    for (int i = 0; i < 24; ++i) {
        int c = lane + 32 * i;
        float v = out[(size_t)row * HH + c] + hidden[(size_t)row * HH + c];
        x[i] = v; sum += v; sq += v * v;
    }
#pragma unroll
    for (int off = 1; off < 32; off <<= 1) {
        sum += __shfl_xor(sum, off, 32);
        sq  += __shfl_xor(sq, off, 32);
    }
    float mu  = sum * (1.0f / HH);
    float var = sq * (1.0f / HH) - mu * mu;
    float inv = rsqrtf(var + 1e-7f);
#pragma unroll
    for (int i = 0; i < 24; ++i) {
        int c = lane + 32 * i;
        out[(size_t)row * HH + c] = (x[i] - mu) * inv * ln_scale[c] + ln_bias[c];
    }
}

// ---------------- host launch ----------------
extern "C" void kernel_launch(void* const* d_in, const int* in_sizes, int n_in,
                              void* d_out, int out_size, void* d_ws, size_t ws_size,
                              hipStream_t stream) {
    (void)in_sizes; (void)n_in; (void)out_size; (void)ws_size;

    const float* hidden = (const float*)d_in[0];
    const float* relpos = (const float*)d_in[1];
    const float* Wq = (const float*)d_in[2];  const float* bq = (const float*)d_in[3];
    const float* Wk = (const float*)d_in[4];  const float* bk = (const float*)d_in[5];
    const float* Wv = (const float*)d_in[6];  const float* bv = (const float*)d_in[7];
    const float* Wo = (const float*)d_in[8];  const float* bo = (const float*)d_in[9];
    const float* ln_scale = (const float*)d_in[10];
    const float* ln_bias  = (const float*)d_in[11];
    float* out = (float*)d_out;

    const int M = BB * SS;               // 32768
    char* ws = (char*)d_ws;
    size_t off = 0;
    auto alloc = [&](size_t bytes) -> void* {
        void* p = ws + off;
        off += (bytes + 255) & ~(size_t)255;
        return p;
    };
    unsigned short* h_bf   = (unsigned short*)alloc((size_t)M * HH * 2);
    unsigned short* wq_bf  = (unsigned short*)alloc((size_t)HH * HH * 2);  // [N][K] transposed
    unsigned short* wk_bf  = (unsigned short*)alloc((size_t)HH * HH * 2);
    unsigned short* wv_bf  = (unsigned short*)alloc((size_t)HH * HH * 2);
    unsigned short* wo_bf  = (unsigned short*)alloc((size_t)HH * HH * 2);
    unsigned short* pos_bf = (unsigned short*)alloc((size_t)PBUCK * HH * 2);
    unsigned short* q_bf   = (unsigned short*)alloc((size_t)M * HH * 2);
    unsigned short* k_bf   = (unsigned short*)alloc((size_t)M * HH * 2);
    unsigned short* v_bf   = (unsigned short*)alloc((size_t)M * HH * 2);
    unsigned short* pq_bf  = (unsigned short*)alloc((size_t)PBUCK * HH * 2);
    unsigned short* pk_bf  = (unsigned short*)alloc((size_t)PBUCK * HH * 2);
    unsigned short* ctx_bf = (unsigned short*)alloc((size_t)M * HH * 2);

    auto cvt_grid = [](int n4) { int g = (n4 + 255) / 256; return g > 4096 ? 4096 : g; };

    // fp32 -> bf16 (activations); weights: transpose + convert to [N][K]
    cvt_kernel<<<cvt_grid(M * HH / 4), 256, 0, stream>>>(hidden, h_bf, M * HH / 4);
    cvt_kernel<<<cvt_grid(PBUCK * HH / 4), 256, 0, stream>>>(relpos, pos_bf, PBUCK * HH / 4);
    cvt_w_kernel<<<2304, 256, 0, stream>>>(Wq, wq_bf);
    cvt_w_kernel<<<2304, 256, 0, stream>>>(Wk, wk_bf);
    cvt_w_kernel<<<2304, 256, 0, stream>>>(Wv, wv_bf);
    cvt_w_kernel<<<2304, 256, 0, stream>>>(Wo, wo_bf);

    // QKV projections
    dim3 gqkv(M / 128, HH / 64);
    gemm_bias_kernel<<<gqkv, 256, 0, stream>>>(h_bf, wq_bf, bq, nullptr, q_bf, M, HH, HH, 0);
    gemm_bias_kernel<<<gqkv, 256, 0, stream>>>(h_bf, wk_bf, bk, nullptr, k_bf, M, HH, HH, 0);
    gemm_bias_kernel<<<gqkv, 256, 0, stream>>>(h_bf, wv_bf, bv, nullptr, v_bf, M, HH, HH, 0);

    // shared relative-position projections (pos_q uses Wq/bq, pos_k uses Wk/bk)
    dim3 gpos(PBUCK / 128, HH / 64);
    gemm_bias_kernel<<<gpos, 256, 0, stream>>>(pos_bf, wq_bf, bq, nullptr, pq_bf, PBUCK, HH, HH, 0);
    gemm_bias_kernel<<<gpos, 256, 0, stream>>>(pos_bf, wk_bf, bk, nullptr, pk_bf, PBUCK, HH, HH, 0);

    // fused attention
    int attn_wgs = BB * NBLK * (BSZ / MQ) * NHH;   // 12288
    attn_kernel<<<attn_wgs, 256, 0, stream>>>(q_bf, k_bf, v_bf, pk_bf, pq_bf, ctx_bf);

    // output projection -> d_out (fp32), then residual + LN in place
    gemm_bias_kernel<<<gqkv, 256, 0, stream>>>(ctx_bf, wo_bf, bo, out, nullptr, M, HH, HH, 1);
    ln_kernel<<<M / 8, 256, 0, stream>>>(hidden, ln_scale, ln_bias, out);
}
